// DynamicPlasticDenseDendritic_31155692765801
// MI455X (gfx1250) — compile-verified, matrix operand order checked by reading, not run on hardware
//
#include <hip/hip_runtime.h>

// ---------------------------------------------------------------------------
// DynamicPlasticDenseDendritic for MI455X (gfx1250, wave32, WMMA)
//   B=1024, D=2048, U=4096, BR=4  ->  GEMM M=1024, K=2048, N=16384
//   Fused: w*sigmoid(delay) on the fly (f32->bf16), bias + branch-softmax
//   combine + connectivity*mask + relu in the epilogue.
// ---------------------------------------------------------------------------

#define B_DIM 1024
#define D_DIM 2048
#define U_DIM 4096
#define BR    4
#define N_DIM (U_DIM * BR)   // 16384
#define HID   32

#define BM 128
#define BN 128
#define BK 32
#define NK (D_DIM / BK)      // 64 K-steps
#define LDSS 72              // LDS row stride in halves (144 B: 16B aligned, bank-spread)

typedef __attribute__((ext_vector_type(16))) __bf16 v16bf;
typedef __attribute__((ext_vector_type(8)))  float  v8f;

__device__ __forceinline__ float frcp_fast(float x) {
    return __builtin_amdgcn_rcpf(x);        // v_rcp_f32, ~1 ULP: plenty for bf16 feed
}
__device__ __forceinline__ float fsigmoid(float x) {
    return frcp_fast(1.0f + __expf(-x));
}

// pack two f32 -> two bf16 in one dword:
//   round-to-nearest (ties away) via +0x8000, then one v_perm_b32 splices the
//   two high halves: dest = {hi.b3, hi.b2, lo.b3, lo.b2}. 3 VALU per pair.
__device__ __forceinline__ unsigned int pk2bf(float lo, float hi) {
    union { float f; unsigned int u; } a, b;
    a.f = lo; b.f = hi;
    return __builtin_amdgcn_perm(b.u + 0x8000u, a.u + 0x8000u, 0x07060302u);
}

__device__ __forceinline__ float quad_swizzle_add(float v) {
    // sum over the 4 lanes of each quad: SWAPX1 then SWAPX2 (group-of-4 mode)
    v += __int_as_float(__builtin_amdgcn_ds_swizzle(__float_as_int(v), 0x80B1));
    v += __int_as_float(__builtin_amdgcn_ds_swizzle(__float_as_int(v), 0x804E));
    return v;
}

// ---------------------------------------------------------------------------
// Kernel 1: gating[b, 0..3] = softmax(inputs[b,:] @ Wg + bg)
// ---------------------------------------------------------------------------
__global__ void gating_kernel(const float* __restrict__ inputs,
                              const float* __restrict__ Wg,
                              const float* __restrict__ bg,
                              float* __restrict__ gate) {
    __shared__ float red[256][4];
    const int b = blockIdx.x;
    const int t = threadIdx.x;
    float4 acc = {0.f, 0.f, 0.f, 0.f};
    const float* row = inputs + (size_t)b * D_DIM;
    for (int d = t; d < D_DIM; d += 256) {
        float  x = row[d];
        float4 w = *(const float4*)(Wg + d * 4);
        acc.x += x * w.x; acc.y += x * w.y; acc.z += x * w.z; acc.w += x * w.w;
    }
    red[t][0] = acc.x; red[t][1] = acc.y; red[t][2] = acc.z; red[t][3] = acc.w;
    __syncthreads();
    for (int s = 128; s > 0; s >>= 1) {
        if (t < s) {
            red[t][0] += red[t + s][0]; red[t][1] += red[t + s][1];
            red[t][2] += red[t + s][2]; red[t][3] += red[t + s][3];
        }
        __syncthreads();
    }
    if (t == 0) {
        float z[4]; float mx = -3.0e38f;
        for (int k = 0; k < 4; k++) { z[k] = red[0][k] + bg[k]; mx = fmaxf(mx, z[k]); }
        float s = 0.f;
        for (int k = 0; k < 4; k++) { z[k] = __expf(z[k] - mx); s += z[k]; }
        float inv = frcp_fast(s);
        for (int k = 0; k < 4; k++) gate[b * 4 + k] = z[k] * inv;
    }
}

// ---------------------------------------------------------------------------
// Kernel 2: connm[u] = sigmoid(relu(avg @ W1 + b1) @ W2 + b2)[u] * mask[u]
// ---------------------------------------------------------------------------
__global__ void conn_kernel(const float* __restrict__ avg,
                            const float* __restrict__ W1, const float* __restrict__ b1,
                            const float* __restrict__ W2, const float* __restrict__ b2,
                            const float* __restrict__ mask,
                            float* __restrict__ connm) {
    __shared__ float part[256];
    __shared__ float h[HID];
    const int t = threadIdx.x;
    const int j = t & 31, g = t >> 5;
    float p = 0.f;
    for (int u = g; u < U_DIM; u += 8) p += avg[u] * W1[u * HID + j];
    part[t] = p;
    __syncthreads();
    if (t < HID) {
        float s = b1[j];
        for (int gg = 0; gg < 8; gg++) s += part[j + 32 * gg];
        h[j] = fmaxf(s, 0.f);
    }
    __syncthreads();
    for (int u = t; u < U_DIM; u += 256) {
        float s = b2[u];
        #pragma unroll
        for (int jj = 0; jj < HID; jj++) s += h[jj] * W2[jj * U_DIM + u];
        connm[u] = fsigmoid(s) * mask[u];
    }
}

// ---------------------------------------------------------------------------
// Kernel 3: main fused GEMM. 128x128 block tile, BK=32, 8 waves of 32x64,
// bf16 WMMA + f32 accumulate, ping-pong double-buffered LDS (1 barrier/step).
// B staged as 2N x 2K per thread: pair-packed dword LDS stores via v_perm_b32,
// float2 coalesced global loads.
// ---------------------------------------------------------------------------
__global__ __launch_bounds__(256)
void dendritic_gemm_kernel(const float* __restrict__ A,     // inputs [B, D]
                           const float* __restrict__ Wt,    // w      [D, N]
                           const float* __restrict__ Dl,    // delay  [D, N]
                           const float* __restrict__ bias,  // b flat [N]
                           const float* __restrict__ gate,  // [B, 4]
                           const float* __restrict__ connm, // [U]
                           float* __restrict__ out) {       // [B, U]
    __shared__ unsigned short sA[2][BM * LDSS];
    __shared__ unsigned short sB[2][BN * LDSS];   // N-major: sB[buf][n*LDSS + k]
    __shared__ float sGate[BM * 4];
    __shared__ float sBias[BN];
    __shared__ float sConn[BN / 4];

    const int t   = threadIdx.x;
    const int blk = blockIdx.x;
    const int bm  = (blk & 7) * BM;     // M-fastest: 8 M-blocks share a w stripe in L2
    const int bn  = (blk >> 3) * BN;

    // stage epilogue operands once
    for (int i = t; i < BM * 4; i += 256) sGate[i] = gate[bm * 4 + i];
    if (t < BN)     sBias[t] = bias[bn + t];
    if (t < BN / 4) sConn[t] = connm[(bn >> 2) + t];

    const int wv   = t >> 5;
    const int lane = t & 31;
    const int wr   = wv >> 1;        // 0..3  -> M offset wr*32
    const int wc   = wv & 1;         // 0..1  -> N offset wc*64
    const int lhi  = lane >> 4;      // half-wave select
    const int l16  = lane & 15;

    v8f acc[2][4];
    #pragma unroll
    for (int i = 0; i < 2; i++)
        #pragma unroll
        for (int j = 0; j < 4; j++) acc[i][j] = (v8f)(0.0f);

    // global staging coordinates
    const int ac4 = t & 7;           // float4 index along K for A
    const int ar  = t >> 3;          // 0..31 A row
    const int bn2 = (t & 63) * 2;    // N offset (pair) for w/delay
    const int bkq = (t >> 6) * 2;    // K base (0,2,4,6); +8*pass covers 32 rows

    float4  av[4];
    float2  wlo[4], whi[4], dlo[4], dhi[4];   // [pass]: rows kk, kk+1

    // ---- prologue: load + stage tile 0 into buffer 0 ----
    #pragma unroll
    for (int p = 0; p < 4; p++) {
        av[p] = *(const float4*)(A + (size_t)(bm + ar + 32 * p) * D_DIM + ac4 * 4);
        const int kk = bkq + 8 * p;
        const size_t o0 = (size_t)kk * N_DIM + bn + bn2;
        wlo[p] = *(const float2*)(Wt + o0);
        whi[p] = *(const float2*)(Wt + o0 + N_DIM);
        dlo[p] = *(const float2*)(Dl + o0);
        dhi[p] = *(const float2*)(Dl + o0 + N_DIM);
    }
    #pragma unroll
    for (int p = 0; p < 4; p++) {
        uint2 pk;
        pk.x = pk2bf(av[p].x, av[p].y);
        pk.y = pk2bf(av[p].z, av[p].w);
        *(uint2*)(&sA[0][(ar + 32 * p) * LDSS + ac4 * 4]) = pk;
        const int kk = bkq + 8 * p;
        *(unsigned int*)(&sB[0][(bn2 + 0) * LDSS + kk]) =
            pk2bf(wlo[p].x * fsigmoid(dlo[p].x), whi[p].x * fsigmoid(dhi[p].x));
        *(unsigned int*)(&sB[0][(bn2 + 1) * LDSS + kk]) =
            pk2bf(wlo[p].y * fsigmoid(dlo[p].y), whi[p].y * fsigmoid(dhi[p].y));
    }
    __syncthreads();

    int buf = 0;
    for (int it = 0; it < NK; ++it) {
        const int ktn = (it + 1) * BK;
        const bool have_next = (ktn < D_DIM);

        // ---- issue next tile's global loads first (covered by the WMMAs) ----
        if (have_next) {
            #pragma unroll
            for (int p = 0; p < 4; p++) {
                av[p] = *(const float4*)(A + (size_t)(bm + ar + 32 * p) * D_DIM + ktn + ac4 * 4);
                const int kk = bkq + 8 * p;
                const size_t o0 = (size_t)(ktn + kk) * N_DIM + bn + bn2;
                wlo[p] = *(const float2*)(Wt + o0);
                whi[p] = *(const float2*)(Wt + o0 + N_DIM);
                dlo[p] = *(const float2*)(Dl + o0);
                dhi[p] = *(const float2*)(Dl + o0 + N_DIM);
            }
            // prime L2 two tiles ahead for the 268MB w/delay stream
            if (ktn + BK < D_DIM) {
                const size_t poff = (size_t)(ktn + BK + bkq) * N_DIM + bn + bn2;
                __builtin_prefetch((const void*)(Wt + poff), 0, 1);
                __builtin_prefetch((const void*)(Dl + poff), 0, 1);
            }
        }

        // ---- fragments from current buffer (ISA 16-bit A/B layout:
        //      lanes 0-15 K0..7 / 16..23, lanes 16-31 K8..15 / 24..31) ----
        union FragU { uint4 q[2]; v16bf v; };
        FragU afr[2], bfr[4];
        #pragma unroll
        for (int mi = 0; mi < 2; mi++) {
            const int row = wr * 32 + mi * 16 + l16;
            const uint4* pb = (const uint4*)(&sA[buf][row * LDSS + lhi * 8]);
            afr[mi].q[0] = pb[0];
            afr[mi].q[1] = pb[2];      // +16 halves
        }
        #pragma unroll
        for (int ni = 0; ni < 4; ni++) {
            const int col = wc * 64 + ni * 16 + l16;
            const uint4* pb = (const uint4*)(&sB[buf][col * LDSS + lhi * 8]);
            bfr[ni].q[0] = pb[0];
            bfr[ni].q[1] = pb[2];
        }

        // ---- 8 WMMAs per wave per K-step ----
        #pragma unroll
        for (int mi = 0; mi < 2; mi++)
            #pragma unroll
            for (int ni = 0; ni < 4; ni++)
                acc[mi][ni] = __builtin_amdgcn_wmma_f32_16x16x32_bf16(
                    false, afr[mi].v, false, bfr[ni].v,
                    (short)0, acc[mi][ni], false, false);

        // ---- stage next tile into the other buffer ----
        if (have_next) {
            const int nb2 = buf ^ 1;
            #pragma unroll
            for (int p = 0; p < 4; p++) {
                uint2 pk;
                pk.x = pk2bf(av[p].x, av[p].y);
                pk.y = pk2bf(av[p].z, av[p].w);
                *(uint2*)(&sA[nb2][(ar + 32 * p) * LDSS + ac4 * 4]) = pk;
                const int kk = bkq + 8 * p;
                *(unsigned int*)(&sB[nb2][(bn2 + 0) * LDSS + kk]) =
                    pk2bf(wlo[p].x * fsigmoid(dlo[p].x), whi[p].x * fsigmoid(dhi[p].x));
                *(unsigned int*)(&sB[nb2][(bn2 + 1) * LDSS + kk]) =
                    pk2bf(wlo[p].y * fsigmoid(dlo[p].y), whi[p].y * fsigmoid(dhi[p].y));
            }
        }
        __syncthreads();
        buf ^= 1;
    }

    // ---- fused epilogue: +bias, x gate (quad reduce over 4 branch cols),
    //      x connectivity*mask, relu, store [B, U] ----
    const int kl = lane & 3;   // branch index k of this lane's column
    #pragma unroll
    for (int mi = 0; mi < 2; mi++) {
        float gv[8];
        #pragma unroll
        for (int r = 0; r < 8; r++) {
            const int lm = wr * 32 + mi * 16 + lhi * 8 + r;
            gv[r] = sGate[lm * 4 + kl];
        }
        #pragma unroll
        for (int ni = 0; ni < 4; ni++) {
            const int ln  = wc * 64 + ni * 16 + l16;
            const float bs = sBias[ln];
            const float cm = sConn[ln >> 2];
            #pragma unroll
            for (int r = 0; r < 8; r++) {
                float v = (acc[mi][ni][r] + bs) * gv[r];
                v = quad_swizzle_add(v);
                if (kl == 0) {
                    const int m = bm + wr * 32 + mi * 16 + lhi * 8 + r;
                    const int u = (bn + ln) >> 2;
                    out[(size_t)m * U_DIM + u] = fmaxf(v * cm, 0.0f);
                }
            }
        }
    }
}

// ---------------------------------------------------------------------------
extern "C" void kernel_launch(void* const* d_in, const int* in_sizes, int n_in,
                              void* d_out, int out_size, void* d_ws, size_t ws_size,
                              hipStream_t stream) {
    const float* inputs = (const float*)d_in[0];
    const float* w      = (const float*)d_in[1];
    const float* b      = (const float*)d_in[2];
    const float* delay  = (const float*)d_in[3];
    const float* Wg     = (const float*)d_in[4];
    const float* bg     = (const float*)d_in[5];
    const float* W1     = (const float*)d_in[6];
    const float* b1     = (const float*)d_in[7];
    const float* W2     = (const float*)d_in[8];
    const float* b2     = (const float*)d_in[9];
    const float* mask   = (const float*)d_in[10];
    const float* avg    = (const float*)d_in[11];
    float* out = (float*)d_out;

    float* gate  = (float*)d_ws;           // B*4 floats
    float* connm = gate + B_DIM * 4;       // U floats

    gating_kernel<<<B_DIM, 256, 0, stream>>>(inputs, Wg, bg, gate);
    conn_kernel<<<1, 256, 0, stream>>>(avg, W1, b1, W2, b2, mask, connm);
    dendritic_gemm_kernel<<<(B_DIM / BM) * (N_DIM / BN), 256, 0, stream>>>(
        inputs, w, delay, b, gate, connm, out);
}
